// Multi_Headed_29145648071316
// MI455X (gfx1250) — compile-verified, matrix operand order checked by reading, not run on hardware
//
#include <hip/hip_runtime.h>
#include <hip/hip_bf16.h>
#include <stdint.h>

// Problem constants (match reference)
#define BB  4
#define SS  2048
#define EE  1024
#define HH  8
#define DHD 128

typedef __attribute__((ext_vector_type(16))) __bf16 v16bf;
typedef __attribute__((ext_vector_type(8)))  __bf16 v8bf;
typedef __attribute__((ext_vector_type(8)))  float  v8f;
typedef __attribute__((ext_vector_type(4)))  int    v4i;

#if __has_builtin(__builtin_amdgcn_global_load_async_to_lds_b128)
#define HAVE_ASYNC_LDS 1
#endif

static __device__ __forceinline__ v8f vzero8() {
  v8f z = {0.f, 0.f, 0.f, 0.f, 0.f, 0.f, 0.f, 0.f};
  return z;
}

static __device__ __forceinline__ v8f wmma_bf16(v16bf a, v16bf b, v8f c) {
  // (neg_a, A, neg_b, B, c_mod, C, reuse_a, reuse_b)
  return __builtin_amdgcn_wmma_f32_16x16x32_bf16(false, a, false, b, (short)0, c,
                                                 false, false);
}

// Fragment loader for the "A pattern" (also B pattern when operand is staged
// transposed): 16 rows x 32 K, row-major, row stride `ld` elements.
// lanes 0-15: row=lane, K in {0..7,16..23}; lanes 16-31: row=lane-16,
// K in {8..15,24..31}. Each lane = two contiguous 16-byte runs -> 2x b128.
static __device__ __forceinline__ v16bf load_frag(const __bf16* base, int ld, int lane) {
  const int half = (lane >> 4) & 1;
  const int r    = lane & 15;
  const __bf16* p = base + r * ld + half * 8;
  const v8bf lo = *(const v8bf*)(p);        // K = k0 .. k0+7
  const v8bf hi = *(const v8bf*)(p + 16);   // K = 16+k0 .. 16+k0+7
  return __builtin_shufflevector(lo, hi, 0, 1, 2, 3, 4, 5, 6, 7,
                                 8, 9, 10, 11, 12, 13, 14, 15);
}

static __device__ __forceinline__ float half_reduce_max(float v) {
  v = fmaxf(v, __shfl_xor(v, 1, 32));
  v = fmaxf(v, __shfl_xor(v, 2, 32));
  v = fmaxf(v, __shfl_xor(v, 4, 32));
  v = fmaxf(v, __shfl_xor(v, 8, 32));
  return v;
}
static __device__ __forceinline__ float half_reduce_sum(float v) {
  v += __shfl_xor(v, 1, 32);
  v += __shfl_xor(v, 2, 32);
  v += __shfl_xor(v, 4, 32);
  v += __shfl_xor(v, 8, 32);
  return v;
}

static __device__ __forceinline__ __bf16 bits_lo(uint32_t v) {
  return __builtin_bit_cast(__bf16, (uint16_t)(v & 0xffffu));
}
static __device__ __forceinline__ __bf16 bits_hi(uint32_t v) {
  return __builtin_bit_cast(__bf16, (uint16_t)(v >> 16));
}

// ---------------------------------------------------------------------------
// Kernel 1: per-head QKV projection, fp32 x -> bf16 Q/K/V [B,H,S,DH]
// grid(S/16, H, B), 256 threads (8 waves; wave = 16-col N-tile of DH=128)
// Weights staged TRANSPOSED in LDS so B-fragments are b128 loads.
// ---------------------------------------------------------------------------
__global__ __launch_bounds__(256)
void qkv_proj_kernel(const float* __restrict__ x,
                     const float* __restrict__ Wq,
                     const float* __restrict__ Wk,
                     const float* __restrict__ Wv,
                     __bf16* __restrict__ Qb,
                     __bf16* __restrict__ Kb,
                     __bf16* __restrict__ Vb) {
  __shared__ __align__(16) __bf16 sX[16][DHD];        // 4 KB, row-major
  __shared__ __align__(16) __bf16 sWt[3][DHD][32];    // 24 KB, transposed slabs

  const int tid  = threadIdx.x;
  const int lane = tid & 31;
  const int wv   = tid >> 5;        // N-tile index (0..7)
  const int s0   = blockIdx.x * 16; // 16-row S tile
  const int h    = blockIdx.y;
  const int b    = blockIdx.z;

  // Stage x tile (fp32 -> bf16): x[b, s0+r, h*DH + c].
  // Per-thread invariant decomposition: c = tid&127, r = tid>>7 + 2i.
  {
    const int c  = tid & (DHD - 1);
    const int r0 = tid >> 7;
    const float* src = x + (size_t)(b * SS + s0 + r0) * EE + h * DHD + c;
#pragma unroll
    for (int i = 0; i < 8; ++i)
      sX[r0 + 2 * i][c] = (__bf16)src[(size_t)(2 * i) * EE];
  }

  v8f acc[3] = {vzero8(), vzero8(), vzero8()};

  for (int kk = 0; kk < 4; ++kk) {          // K-dim: 4 chunks of 32
    __syncthreads();
    // Stage Wq/Wk/Wv K-slabs transposed: sWt[g][n][k] = W[g][h][kk*32+k][n].
    // g unrolled -> pointer select folds; n = tid&127 invariant, k = tid>>7+2i.
    {
      const int n  = tid & (DHD - 1);
      const int k0 = tid >> 7;
#pragma unroll
      for (int g = 0; g < 3; ++g) {
        const float* Wp = (g == 0) ? Wq : ((g == 1) ? Wk : Wv);
        const float* src =
            Wp + (size_t)h * DHD * DHD + (size_t)(kk * 32 + k0) * DHD + n;
        __bf16* dst = &sWt[g][n][k0];
#pragma unroll
        for (int i = 0; i < 16; ++i)
          dst[2 * i] = (__bf16)src[(size_t)(2 * i) * DHD];
      }
    }
    __syncthreads();

    const v16bf a = load_frag(&sX[0][kk * 32], DHD, lane);
#pragma unroll
    for (int g = 0; g < 3; ++g) {
      const v16bf bf = load_frag(&sWt[g][wv * 16][0], 32, lane);
      acc[g] = wmma_bf16(a, bf, acc[g]);
    }
  }

  // Store D tiles: VGPR i, lane-half -> row i (+8), col = lane&15
  const int half = (lane >> 4) & 1;
  const int col  = lane & 15;
  const size_t obase =
      ((size_t)(b * HH + h) * SS + s0 + 8 * half) * DHD + wv * 16 + col;
#pragma unroll
  for (int i = 0; i < 8; ++i) {
    Qb[obase + (size_t)i * DHD] = (__bf16)acc[0][i];
    Kb[obase + (size_t)i * DHD] = (__bf16)acc[1][i];
    Vb[obase + (size_t)i * DHD] = (__bf16)acc[2][i];
  }
}

// ---------------------------------------------------------------------------
// Kernel 2: causal flash attention per (b,h), 128 q-rows per block.
// Each wave owns 16 q-rows and the full DH=128 output (8 f32 accum tiles).
// Scale is 1/sqrt(S) per the reference (NOT 1/sqrt(DH)).
// grid(S/128, H, B), 256 threads.
// ---------------------------------------------------------------------------
__global__ __launch_bounds__(256)
void flash_attn_kernel(const __bf16* __restrict__ Qb,
                       const __bf16* __restrict__ Kb,
                       const __bf16* __restrict__ Vb,
                       __bf16* __restrict__ Ab) {
  __shared__ __align__(16) __bf16 sK[32][DHD];     // 8 KB, row-major (keys x d)
  __shared__ __align__(16) __bf16 sVt[DHD][32];    // 8 KB, transposed (d x keys)
  __shared__ __align__(16) __bf16 sP[8][16][32];   // 8 KB (per-wave P tiles)

  const int tid  = threadIdx.x;
  const int lane = tid & 31;
  const int wv   = tid >> 5;
  const int q0   = blockIdx.x * 128;
  const int h    = blockIdx.y;
  const int b    = blockIdx.z;
  const int qw   = q0 + wv * 16;    // this wave's first q row
  const int half = (lane >> 4) & 1;
  const int col  = lane & 15;

  const size_t headBase = (size_t)(b * HH + h) * SS * DHD;
  const __bf16* Qh = Qb + headBase;
  const __bf16* Kh = Kb + headBase;
  const __bf16* Vh = Vb + headBase;

  // Resident Q fragments: 16 x 128 = 4 A-frags (global b128 loads)
  v16bf qf[4];
#pragma unroll
  for (int c = 0; c < 4; ++c)
    qf[c] = load_frag(Qh + (size_t)qw * DHD + c * 32, DHD, lane);

  v8f O[8];
#pragma unroll
  for (int t = 0; t < 8; ++t) O[t] = vzero8();
  float rmax[8], rsum[8];
#pragma unroll
  for (int i = 0; i < 8; ++i) { rmax[i] = -__builtin_inff(); rsum[i] = 0.f; }

  const float kScale = 0.022097086912079608f;   // 1/sqrt(2048)
  const int nkb = q0 / 32 + 4;                  // causal bound for this block

  // Per-thread invariants for the V transpose staging:
  // e = tid + 256r -> d = (tid&63)*2 constant, k = (tid>>6) + 4r.
  const int vD  = (tid & 63) * 2;
  const int vK0 = tid >> 6;

  for (int kb = 0; kb < nkb; ++kb) {
    const int kBase = kb * 32;

    if (kb + 1 < nkb) {  // overlap next K/V block fetch (global_prefetch_b8)
      __builtin_prefetch(Kh + (size_t)(kBase + 32) * DHD, 0, 1);
      __builtin_prefetch(Vh + (size_t)(kBase + 32) * DHD, 0, 1);
    }

    // --- K tile: async DMA to LDS if available, else vector copy ------------
    {
      const __bf16* gsrc = Kh + (size_t)kBase * DHD;
      __bf16* lbase = &sK[0][0];
#ifdef HAVE_ASYNC_LDS
#pragma unroll
      for (int r = 0; r < 2; ++r) {
        const int c = tid + r * 256;                 // 512 x 16B chunks
        __builtin_amdgcn_global_load_async_to_lds_b128(
            (v4i*)(gsrc + c * 8), (v4i*)(lbase + c * 8), 0, 0);
      }
#else
      const uint4* src = (const uint4*)gsrc;
      uint4* dst = (uint4*)lbase;
#pragma unroll
      for (int r = 0; r < 2; ++r) dst[tid + r * 256] = src[tid + r * 256];
#endif
    }

    // --- V tile: staged TRANSPOSED (register transpose, b32 global reads) ---
    {
      const uint32_t* Vsrc = (const uint32_t*)(Vh + (size_t)kBase * DHD) + tid;
      __bf16* d0 = &sVt[vD][vK0];
      __bf16* d1 = &sVt[vD + 1][vK0];
#pragma unroll
      for (int r = 0; r < 8; ++r) {
        const uint32_t v = Vsrc[r * 256];
        d0[4 * r] = bits_lo(v);            // k advances by 4 per r
        d1[4 * r] = bits_hi(v);
      }
    }

#ifdef HAVE_ASYNC_LDS
#if __has_builtin(__builtin_amdgcn_s_wait_asynccnt)
    __builtin_amdgcn_s_wait_asynccnt(0);
#else
    asm volatile("s_wait_asynccnt 0" ::: "memory");
#endif
#endif
    __syncthreads();

    // Scores: S = Q @ K^T. B[k][n] = K[n][k] -> A-pattern read on row-major sK.
    v8f sa = vzero8(), sb = vzero8();
#pragma unroll
    for (int c = 0; c < 4; ++c) {
      const v16bf bk0 = load_frag(&sK[0][c * 32],  DHD, lane);  // keys 0..15
      const v16bf bk1 = load_frag(&sK[16][c * 32], DHD, lane);  // keys 16..31
      sa = wmma_bf16(qf[c], bk0, sa);
      sb = wmma_bf16(qf[c], bk1, sb);
    }

    // Online softmax (per row i -> q row qw + i + 8*half)
    float pa[8], pb[8];
#pragma unroll
    for (int i = 0; i < 8; ++i) {
      const int qrow = qw + i + 8 * half;
      const float xa = (kBase + col      <= qrow) ? sa[i] * kScale : -__builtin_inff();
      const float xb = (kBase + 16 + col <= qrow) ? sb[i] * kScale : -__builtin_inff();
      float mx = half_reduce_max(fmaxf(xa, xb));
      const float mnew  = fmaxf(rmax[i], mx);
      const float alpha = __expf(rmax[i] - mnew);
      const float ea = __expf(xa - mnew);
      const float eb = __expf(xb - mnew);
      rsum[i] = rsum[i] * alpha + half_reduce_sum(ea + eb);
      rmax[i] = mnew;
      pa[i] = ea;
      pb[i] = eb;
#pragma unroll
      for (int t = 0; t < 8; ++t) O[t][i] *= alpha;   // rescale output rows
    }

    // P (D-layout) -> LDS -> A-layout fragment (intra-wave round trip)
    {
      __bf16* p0 = &sP[wv][8 * half][col];
#pragma unroll
      for (int i = 0; i < 8; ++i) {
        p0[i * 32]      = (__bf16)pa[i];
        p0[i * 32 + 16] = (__bf16)pb[i];
      }
    }
    asm volatile("s_wait_dscnt 0" ::: "memory");   // intra-wave DS RAW fence
    const v16bf pf = load_frag(&sP[wv][0][0], 32, lane);

    // O += P @ V  (8 N-tiles cover DH=128); B-frag = A-pattern on transposed sVt
#pragma unroll
    for (int t = 0; t < 8; ++t) {
      const v16bf bv = load_frag(&sVt[t * 16][0], 32, lane);
      O[t] = wmma_bf16(pf, bv, O[t]);
    }
    __syncthreads();   // protect sK/sVt before next iteration overwrites
  }

  // Normalize and emit attn in [B, S, H*DH] (bqhd order) as bf16
  float inv[8];
#pragma unroll
  for (int i = 0; i < 8; ++i) inv[i] = 1.0f / rsum[i];

  __bf16* obase = Ab + (size_t)(b * SS + qw + 8 * half) * EE + h * DHD + col;
#pragma unroll
  for (int t = 0; t < 8; ++t) {
#pragma unroll
    for (int i = 0; i < 8; ++i) {
      obase[(size_t)i * EE + t * 16] = (__bf16)(O[t][i] * inv[i]);
    }
  }
}

// ---------------------------------------------------------------------------
// Kernel 3: output projection  out[8192,1024] = Ab @ W_out + b_out (fp32 out)
// grid(B*S/16, E/128), 256 threads (8 waves; wave = 16-col N-tile)
// ---------------------------------------------------------------------------
__global__ __launch_bounds__(256)
void out_proj_kernel(const __bf16* __restrict__ Ab,
                     const float* __restrict__ Wout,
                     const float* __restrict__ bout,
                     float* __restrict__ out) {
  __shared__ __align__(16) __bf16 sA[16][32];      // 1 KB, row-major
  __shared__ __align__(16) __bf16 sWt[DHD][32];    // 8 KB, transposed

  const int tid  = threadIdx.x;
  const int lane = tid & 31;
  const int wv   = tid >> 5;
  const int r0   = blockIdx.x * 16;            // 16-row tile
  const int cg   = blockIdx.y;                 // 128-col group
  const int c0   = cg * 128 + wv * 16;

  // Per-thread invariants:
  //  sA copy: 512 elems, 2/thread: r = tid>>5 + 8i, c = tid&31
  //  sWt copy: 4096 elems, 16/thread: n = tid&127, k = tid>>7 + 2i
  const int ar = tid >> 5;
  const int ac = tid & 31;
  const int wn = tid & (DHD - 1);
  const int wk = tid >> 7;

  v8f acc = vzero8();
  for (int kk = 0; kk < EE / 32; ++kk) {
    __syncthreads();
    {
      const __bf16* src = Ab + (size_t)(r0 + ar) * EE + kk * 32 + ac;
#pragma unroll
      for (int i = 0; i < 2; ++i) sA[ar + 8 * i][ac] = src[(size_t)(8 * i) * EE];
    }
    // sWt[n][k] = Wout[kk*32+k][cg*128+n]   (coalesced reads over n)
    {
      const float* src = Wout + (size_t)(kk * 32 + wk) * EE + cg * 128 + wn;
      __bf16* dst = &sWt[wn][wk];
#pragma unroll
      for (int i = 0; i < 16; ++i) dst[2 * i] = (__bf16)src[(size_t)(2 * i) * EE];
    }
    __syncthreads();
    const v16bf a  = load_frag(&sA[0][0], 32, lane);
    const v16bf bf = load_frag(&sWt[wv * 16][0], 32, lane);
    acc = wmma_bf16(a, bf, acc);
  }

  const int half = (lane >> 4) & 1;
  const int col  = lane & 15;
  const float bias = bout[c0 + col];
  float* obase = out + (size_t)(r0 + 8 * half) * EE + c0 + col;
#pragma unroll
  for (int i = 0; i < 8; ++i) obase[(size_t)i * EE] = acc[i] + bias;
}

// ---------------------------------------------------------------------------
extern "C" void kernel_launch(void* const* d_in, const int* in_sizes, int n_in,
                              void* d_out, int out_size, void* d_ws, size_t ws_size,
                              hipStream_t stream) {
  // setup_inputs() order: x, Wk, Wv, Wq, W_out, b_out
  const float* x    = (const float*)d_in[0];
  const float* Wk   = (const float*)d_in[1];
  const float* Wv   = (const float*)d_in[2];
  const float* Wq   = (const float*)d_in[3];
  const float* Wout = (const float*)d_in[4];
  const float* bout = (const float*)d_in[5];
  float* out = (float*)d_out;

  // Workspace: bf16 Q,K,V [B,H,S,DH] + bf16 attn [B*S, E]  (4 x 16 MB = 64 MB)
  const size_t perTensor = (size_t)BB * HH * SS * DHD;
  __bf16* Qb = (__bf16*)d_ws;
  __bf16* Kb = Qb + perTensor;
  __bf16* Vb = Kb + perTensor;
  __bf16* Ab = Vb + perTensor;

  qkv_proj_kernel<<<dim3(SS / 16, HH, BB), 256, 0, stream>>>(x, Wq, Wk, Wv, Qb, Kb, Vb);
  flash_attn_kernel<<<dim3(SS / 128, HH, BB), 256, 0, stream>>>(Qb, Kb, Vb, Ab);
  out_proj_kernel<<<dim3((BB * SS) / 16, EE / 128), 256, 0, stream>>>(Ab, Wout, bout, out);
}
